// ModernHopfieldScaled_43224550867320
// MI455X (gfx1250) — matrix operand order repaired and checked by reading, not compiled, not run on hardware
//
#include <hip/hip_runtime.h>
#include <hip/hip_bf16.h>

typedef __attribute__((ext_vector_type(16))) __bf16 v16bf;
typedef __attribute__((ext_vector_type(8)))  float  v8f;

#define NB  65536
#define ND  1024
#define NP  240
#define NLV 4

#define OUT_ELEMS ((size_t)NB * (size_t)ND)      // 67,108,864 floats, then attn [L,B,240]

// workspace layout (bytes)
#define WS_SCAL_OFF 0        // float[32]: [0..3] scales, [4..7] betas, [8..15] b8, [16..19] 1/scale
#define WS_W8_OFF   1024     // bf16 W8frag[32 kt][32 lane][16]  = 32 KB  (B-fragment layout)
#define WS_VP_OFF   65536    // bf16 VpFrag[32 lkt][64 nt][32 lane][16] = 2 MB (B-fragment layout)

static __device__ __forceinline__ v16bf zero16bf() {
  v16bf z;
#pragma unroll
  for (int i = 0; i < 16; ++i) z[i] = (__bf16)0.0f;
  return z;
}

// ---------------------------------------------------------------------------
// K0: scalars (exp/clip), fused bias b8 = e8_w@q_b + e8_b, and
//     W8[n][k] = sum_t e8_w[n][t] * q_w[t][k] stored as WMMA B-fragments (bf16).
// grid 64 x 256  ->  16384 threads = 32 ktiles * 32 lanes * 16 elems
// ---------------------------------------------------------------------------
__global__ __launch_bounds__(256) void k0_prep(
    const float* __restrict__ q_w, const float* __restrict__ q_b,
    const float* __restrict__ e8_w, const float* __restrict__ e8_b,
    const float* __restrict__ lls, const float* __restrict__ lbt,
    float* __restrict__ scal, __bf16* __restrict__ w8frag) {
  int t    = blockIdx.x * blockDim.x + threadIdx.x;
  int kt   = t >> 9;
  int lane = (t >> 4) & 31;
  int i    = t & 15;
  int k = kt * 32 + (lane >> 4) * 16 + i;   // B-frag: k = ktile*32 + half*16 + i
  int n = lane & 15;                        // B-frag: n = lane&15 (cols 8..15 are zero pad)
  float acc = 0.0f;
  if (n < 8) {
    const float* er = e8_w + (size_t)n * ND;
#pragma unroll 4
    for (int tt = 0; tt < ND; ++tt) acc = fmaf(er[tt], q_w[(size_t)tt * ND + k], acc);
  }
  w8frag[((size_t)(kt * 32 + lane)) * 16 + i] = (__bf16)acc;

  if (blockIdx.x == 0) {
    int tid = threadIdx.x;
    if (tid < 4) {
      float s = fminf(fmaxf(__expf(lls[tid]), 0.1f), 10000.0f);
      scal[tid] = s;
      scal[16 + tid] = 1.0f / s;
    } else if (tid < 8) {
      scal[tid] = fminf(fmaxf(__expf(lbt[tid - 4]), 0.1f), 100.0f);
    } else if (tid < 16) {
      int nn = tid - 8;
      float a2 = e8_b[nn];
      const float* er = e8_w + (size_t)nn * ND;
      for (int tt = 0; tt < ND; ++tt) a2 = fmaf(er[tt], q_b[tt], a2);
      scal[tid] = a2;
    }
  }
}

// ---------------------------------------------------------------------------
// K1: VpFrag = (level_values @ out_w.T) / scale[l], bf16 WMMA GEMM 960x1024x1024.
// One wave per 16x16 tile; 60 M-tiles (15 per level, no level crossing) x 64 N-tiles.
// grid 960 x 128 (4 waves/block).
// ---------------------------------------------------------------------------
__global__ __launch_bounds__(128) void k1_vp(
    const float* __restrict__ lv, const float* __restrict__ out_w,
    const float* __restrict__ scal, __bf16* __restrict__ vpfrag) {
  int w    = blockIdx.x * 4 + (threadIdx.x >> 5);   // 0..3839
  int lane = threadIdx.x & 31;
  int mt = w >> 6;                                  // 0..59
  int nt = w & 63;
  int l  = mt / 15;
  int r0 = mt * 16;
  int half = lane >> 4;
  int mrow = lane & 15;
  int ncol = nt * 16 + mrow;

  v8f C = {};
  const float* arow = lv + (size_t)(r0 + mrow) * ND;
  const float* brow = out_w + (size_t)ncol * ND;    // B[k][n] = out_w[n][k]
#pragma unroll 4
  for (int kt = 0; kt < 32; ++kt) {
    int kb = kt * 32;
    v16bf a, b;
#pragma unroll
    for (int v = 0; v < 8; ++v) {                   // A 16-bit frag k pattern
      int kk = kb + half * 16 + (v << 1);
      float2 f = *(const float2*)(arow + kk);
      a[2 * v]     = (__bf16)f.x;
      a[2 * v + 1] = (__bf16)f.y;
    }
    int kbb = kb + half * 16;                       // B frag: k = half*16 + i (contiguous)
#pragma unroll
    for (int q4 = 0; q4 < 4; ++q4) {
      float4 f4 = *(const float4*)(brow + kbb + q4 * 4);
      b[4 * q4]     = (__bf16)f4.x;
      b[4 * q4 + 1] = (__bf16)f4.y;
      b[4 * q4 + 2] = (__bf16)f4.z;
      b[4 * q4 + 3] = (__bf16)f4.w;
    }
    C = __builtin_amdgcn_wmma_f32_16x16x32_bf16(false, a, false, b, (short)0, C, false, false);
  }
  float invs = scal[16 + l];
#pragma unroll
  for (int rr = 0; rr < 8; ++rr) {
    int j     = (mt % 15) * 16 + rr + 8 * half;     // row within level, 0..239
    int kt3   = j >> 5;
    int lane2 = ((j >> 4) & 1) * 16 + mrow;
    int ii    = j & 15;
    size_t idx = (((size_t)(l * 8 + kt3) * 64 + nt) * 32 + lane2) * 16 + ii;
    vpfrag[idx] = (__bf16)(C[rr] * invs);
  }
}

// ---------------------------------------------------------------------------
// K2: per-wave 16 rows: q_e8 = query @ W8.T + b8 (bf16 WMMA, K=1024), normalize
// to sqrt(2), then 4-level Hopfield loop: sim/softmax/argmax/residual.
// Lanes parallel over the 240 codewords; attn written f32 into d_out.
// grid 1024 x 128 (4 waves/block).
// ---------------------------------------------------------------------------
__global__ __launch_bounds__(128) void k2_hopfield(
    const float* __restrict__ query, const __bf16* __restrict__ w8frag,
    const float* __restrict__ scal, const float* __restrict__ cbg,
    float* __restrict__ outp) {
  __shared__ __align__(16) float cb[NP * 8];
  __shared__ __align__(16) float res[4][16][8];

  int tid  = threadIdx.x;
  int wv   = tid >> 5;
  int lane = tid & 31;
  int half = lane >> 4;
  int mrow = lane & 15;
  int rowBase = (blockIdx.x * 4 + wv) * 16;

  for (int idx = tid; idx < NP * 8; idx += blockDim.x) cb[idx] = cbg[idx];

  float sc[NLV], bt[NLV];
#pragma unroll
  for (int l = 0; l < NLV; ++l) { sc[l] = scal[l]; bt[l] = scal[4 + l]; }
  float iv0 = scal[16], iv1 = scal[17], iv2 = scal[18], iv3 = scal[19];
  float ivv[NLV] = {iv0, iv1, iv2, iv3};
  float b8n = (mrow < 8) ? scal[8 + mrow] : 0.0f;

  // ---- stage 1: q_e8 via WMMA bf16 ----
  v8f C = {};
  const float* qrow = query + (size_t)(rowBase + mrow) * ND;
#pragma unroll 4
  for (int kt = 0; kt < 32; ++kt) {
    int kb = kt * 32;
    v16bf a;
#pragma unroll
    for (int v = 0; v < 8; ++v) {
      int kk = kb + half * 16 + (v << 1);
      float2 f = *(const float2*)(qrow + kk);
      a[2 * v]     = (__bf16)f.x;
      a[2 * v + 1] = (__bf16)f.y;
    }
    v16bf b = *(const v16bf*)(w8frag + ((size_t)(kt * 32 + lane)) * 16);
    C = __builtin_amdgcn_wmma_f32_16x16x32_bf16(false, a, false, b, (short)0, C, false, false);
  }
  // row-wise norm over n=0..7 (8-lane shuffle groups), residual -> LDS
#pragma unroll
  for (int rr = 0; rr < 8; ++rr) {
    float v = C[rr] + b8n;          // cols >= 8 are exactly 0 (zero-padded B)
    float s = v * v;
    s += __shfl_xor(s, 1, 8);
    s += __shfl_xor(s, 2, 8);
    s += __shfl_xor(s, 4, 8);
    float scl = 1.41421356237f / fmaxf(sqrtf(s), 1e-12f);
    if (mrow < 8) res[wv][rr + 8 * half][mrow] = v * scl;
  }
  __syncthreads();

  // ---- stage 2: level loop ----
  const float LOG2E = 1.4426950408889634f;
  for (int level = 0; level < NLV; ++level) {
    float scale = sc[level], beta = bt[level], invs = ivv[level];
    for (int ri = 0; ri < 16; ++ri) {
      int row = rowBase + ri;
      float4 ra = *(const float4*)&res[wv][ri][0];
      float4 rb = *(const float4*)&res[wv][ri][4];
      float simv[8];
      float lmax = -3.0e38f;
      int   lbest = 0;
#pragma unroll
      for (int c = 0; c < 8; ++c) {
        int j = lane + 32 * c;
        float s = -3.0e38f;
        if (j < NP) {
          const float* cr = &cb[j * 8];
          float4 c0 = *(const float4*)cr;
          float4 c1 = *(const float4*)(cr + 4);
          s = ra.x * c0.x + ra.y * c0.y + ra.z * c0.z + ra.w * c0.w +
              rb.x * c1.x + rb.y * c1.y + rb.z * c1.z + rb.w * c1.w;
          s *= invs;
          if (s > lmax) { lmax = s; lbest = j; }
        }
        simv[c] = s;
      }
      // wave-wide argmax (lowest index wins ties, matching jnp.argmax)
#pragma unroll
      for (int off = 16; off >= 1; off >>= 1) {
        float om = __shfl_xor(lmax, off);
        int   oj = __shfl_xor(lbest, off);
        if (om > lmax || (om == lmax && oj < lbest)) { lmax = om; lbest = oj; }
      }
      float lsum = 0.0f;
      float ev[8];
#pragma unroll
      for (int c = 0; c < 8; ++c) {
        float e = 0.0f;
        if (lane + 32 * c < NP) e = exp2f((simv[c] - lmax) * beta * LOG2E);
        ev[c] = e;
        lsum += e;
      }
#pragma unroll
      for (int off = 16; off >= 1; off >>= 1) lsum += __shfl_xor(lsum, off);
      float rinv = 1.0f / lsum;
      float* ao = outp + OUT_ELEMS + ((size_t)level * NB + row) * NP;
#pragma unroll
      for (int c = 0; c < 8; ++c) {
        int j = lane + 32 * c;
        if (j < NP) ao[j] = ev[c] * rinv;
      }
      if (level < NLV - 1 && lane < 8) {
        res[wv][ri][lane] -= cb[lbest * 8 + lane] * scale;   // residual -= cb[idx]*scale
      }
    }
    __syncthreads();
  }
}

// ---------------------------------------------------------------------------
// K3: out = A[B,960] @ Vp[960,1024] + out_b, A = attentions (f32 in d_out, cvt
// to bf16 on load). K per level padded 240->256 (kt==7: A lanes 16..31 zeroed;
// B lanes 16..31 masked). 16x128 tile per wave, 8 accumulators.
// grid 8192 x 128 (4 waves/block) -> 32768 waves.
// ---------------------------------------------------------------------------
__global__ __launch_bounds__(128) void k3_out(
    const float* __restrict__ attn, const __bf16* __restrict__ vpfrag,
    const float* __restrict__ out_b, float* __restrict__ outp) {
  int w    = blockIdx.x * 4 + (threadIdx.x >> 5);   // 0..32767
  int lane = threadIdx.x & 31;
  int half = lane >> 4;
  int mrow = lane & 15;
  int mt = w >> 3;                                  // 0..4095
  int ng = w & 7;
  int rowBase = mt * 16;
  int col0 = ng * 128;

  v8f zf = {};
  v8f C[8];
#pragma unroll
  for (int t = 0; t < 8; ++t) C[t] = zf;

#pragma unroll
  for (int level = 0; level < NLV; ++level) {
    const float* arow = attn + ((size_t)level * NB + rowBase + mrow) * NP;
    __builtin_prefetch(arow, 0, 0);                 // global_prefetch_b8 for A stream
#pragma unroll
    for (int kt = 0; kt < 8; ++kt) {
      v16bf a;
      int jb = kt * 32;
#pragma unroll
      for (int v = 0; v < 8; ++v) {
        if (kt == 7 && half == 1) {                 // j >= 240: zero pad
          a[2 * v]     = (__bf16)0.0f;
          a[2 * v + 1] = (__bf16)0.0f;
        } else {
          int jj = jb + half * 16 + (v << 1);
          float2 f = *(const float2*)(arow + jj);
          a[2 * v]     = (__bf16)f.x;
          a[2 * v + 1] = (__bf16)f.y;
        }
      }
      const __bf16* bbase =
          vpfrag + (((size_t)(level * 8 + kt) * 64) + (col0 >> 4)) * 32 * 16;
#pragma unroll
      for (int t = 0; t < 8; ++t) {
        v16bf b;
        if (kt == 7 && half == 1) b = zero16bf();   // k rows 240..255 never written
        else b = *(const v16bf*)(bbase + ((size_t)t * 32 + lane) * 16);
        C[t] = __builtin_amdgcn_wmma_f32_16x16x32_bf16(false, a, false, b, (short)0,
                                                       C[t], false, false);
      }
    }
  }
  // epilogue: + out_b, store f32
#pragma unroll
  for (int t = 0; t < 8; ++t) {
    int ncol = col0 + t * 16 + mrow;
    float bias = out_b[ncol];
    float* orow = outp + (size_t)rowBase * ND + ncol;
#pragma unroll
    for (int rr = 0; rr < 8; ++rr) {
      orow[(size_t)(rr + 8 * half) * ND] = C[t][rr] + bias;
    }
  }
}

extern "C" void kernel_launch(void* const* d_in, const int* in_sizes, int n_in,
                              void* d_out, int out_size, void* d_ws, size_t ws_size,
                              hipStream_t stream) {
  const float* query = (const float*)d_in[0];
  const float* q_w   = (const float*)d_in[1];
  const float* q_b   = (const float*)d_in[2];
  const float* e8_w  = (const float*)d_in[3];
  const float* e8_b  = (const float*)d_in[4];
  const float* out_w = (const float*)d_in[5];
  const float* out_b = (const float*)d_in[6];
  const float* lls   = (const float*)d_in[7];
  const float* lbt   = (const float*)d_in[8];
  const float* lv    = (const float*)d_in[9];
  const float* cbg   = (const float*)d_in[10];
  (void)in_sizes; (void)n_in; (void)out_size; (void)ws_size;

  float*   outp   = (float*)d_out;
  float*   scal   = (float*)((char*)d_ws + WS_SCAL_OFF);
  __bf16*  w8frag = (__bf16*)((char*)d_ws + WS_W8_OFF);
  __bf16*  vpfrag = (__bf16*)((char*)d_ws + WS_VP_OFF);

  k0_prep<<<64, 256, 0, stream>>>(q_w, q_b, e8_w, e8_b, lls, lbt, scal, w8frag);
  k1_vp<<<960, 128, 0, stream>>>(lv, out_w, scal, vpfrag);
  k2_hopfield<<<1024, 128, 0, stream>>>(query, w8frag, scal, cbg, outp);
  k3_out<<<8192, 128, 0, stream>>>(outp + OUT_ELEMS, vpfrag, out_b, outp);
}
